// GATModelSAG_20993800143361
// MI455X (gfx1250) — matrix operand order repaired and checked by reading, not compiled, not run on hardware
//
#include <hip/hip_runtime.h>
#include <hip/hip_bf16.h>

// ---------------------------------------------------------------------------
// GATv2 x2 + SAGPooling for MI455X (gfx1250, wave32).
// Dense GEMMs via v_wmma_f32_16x16x32_bf16 (bf16 compute, fp32 accumulate),
// fully unrolled over K, two 16x16 output tiles per wave sharing B fragments.
// Edge softmax via 3 bandwidth-bound passes with float atomics.
// ---------------------------------------------------------------------------

#define NNODES 32768
#define NEDGES 262144
#define NEA    (NEDGES + NNODES)   // edges + self loops = 294912
#define F1C    256                 // hidden width (heads*ch)
#define NB     128                 // graphs
#define NGP    256                 // nodes per graph
#define KEEP   128                 // top-k kept per graph

typedef __attribute__((ext_vector_type(16))) __bf16 v16bf;
typedef __attribute__((ext_vector_type(8)))  float  v8f;

static __device__ __forceinline__ void atomicMaxF(float* addr, float val) {
    unsigned* ua = (unsigned*)addr;
    unsigned old = __float_as_uint(*addr);
    while (true) {
        float cur = __uint_as_float(old);
        if (cur >= val) break;
        unsigned assumed = old;
        old = atomicCAS(ua, assumed, __float_as_uint(val));
        if (old == assumed) break;
    }
}

// A-fragment loader: 16x32 bf16 A tile, lane layout per ISA 7.12.2:
// lane(l) row = l&15; k-base kb = 8*(l>>4); elems j=0..7 -> k=kb+j,
// j=8..15 -> k=kb+16+(j-8)  (two contiguous 8-float runs -> 4x float4).
static __device__ __forceinline__ v16bf load_a_bf16(const float* __restrict__ p) {
    float4 a0 = *(const float4*)(p);
    float4 a1 = *(const float4*)(p + 4);
    float4 a2 = *(const float4*)(p + 16);
    float4 a3 = *(const float4*)(p + 20);
    float af[16] = {a0.x, a0.y, a0.z, a0.w, a1.x, a1.y, a1.z, a1.w,
                    a2.x, a2.y, a2.z, a2.w, a3.x, a3.y, a3.z, a3.w};
    v16bf v;
    #pragma unroll
    for (int j = 0; j < 16; ++j) v[j] = (__bf16)af[j];   // hw cvt, not manual RNE
    return v;
}

// ---------------------------------------------------------------------------
// fill
// ---------------------------------------------------------------------------
__global__ void k_fill(float* __restrict__ p, float v, size_t n) {
    size_t t = (size_t)blockIdx.x * blockDim.x + threadIdx.x;
    if (t < n) p[t] = v;
}

// ---------------------------------------------------------------------------
// self-loop attr = mean of incoming edge attrs   (fill_value='mean')
// ---------------------------------------------------------------------------
__global__ void k_loop_accum(const int* __restrict__ dst,
                             const float* __restrict__ eattr,
                             float* __restrict__ loopsum,
                             float* __restrict__ cnt, int E) {
    int t = blockIdx.x * blockDim.x + threadIdx.x;
    if (t >= E * 32) return;
    int e = t >> 5, j = t & 31;
    int d = dst[e];
    atomicAdd(&loopsum[(size_t)d * 32 + j], eattr[(size_t)e * 32 + j]);
    if (j == 0) atomicAdd(&cnt[d], 1.0f);
}

__global__ void k_loop_div(float* __restrict__ loopattr,
                           const float* __restrict__ cnt, int n) {
    int t = blockIdx.x * blockDim.x + threadIdx.x;
    if (t >= n * 32) return;
    float c = fmaxf(cnt[t >> 5], 1.0f);
    loopattr[t] = loopattr[t] / c;
}

__global__ void k_build_edges(const int* __restrict__ ei,
                              int* __restrict__ srcA, int* __restrict__ dstA,
                              int E, int EA) {
    int t = blockIdx.x * blockDim.x + threadIdx.x;
    if (t >= EA) return;
    if (t < E) { srcA[t] = ei[t]; dstA[t] = ei[E + t]; }
    else       { srcA[t] = t - E; dstA[t] = t - E; }
}

// ---------------------------------------------------------------------------
// WMMA GEMM:  C[M x Nc] = A[M x KD] @ W[KD x Nc] + bias
// 8 waves/block; each wave owns a 32x16 output slab = two 16x16 WMMA tiles
// sharing the B fragment. K fully unrolled (KD is 64 or 256).
// B layout: lane = column n, K order as A: j<8 -> k=kb+j, j>=8 -> k=kb+16+(j-8), kb=8*(lane>=16).
// C/D: 8 VGPRs, M = r + 8*(lane>=16), N = lane&15.
// ---------------------------------------------------------------------------
template <int KD>
__global__ __launch_bounds__(256)
void k_wmma_gemm_bias(const float* __restrict__ A, const float* __restrict__ W,
                      const float* __restrict__ bias, float* __restrict__ C,
                      int M, int Nc) {
    const int lane = threadIdx.x & 31;
    const int wave = threadIdx.x >> 5;
    const int row0 = blockIdx.y * 32;
    const int col0 = blockIdx.x * 128 + wave * 16;
    if (row0 >= M || col0 >= Nc) return;

    const int m    = lane & 15;
    const int half = lane >> 4;        // 0 or 1
    const int kb   = half * 8;         // A k-base within 16-wide half
    const int n    = lane & 15;        // B/C column
    const int bkb  = half * 16;        // unused; B loads use kb

    v8f acc0 = {}, acc1 = {};
    const float* arow0 = A + (size_t)(row0 + m) * KD;
    const float* arow1 = A + (size_t)(row0 + 16 + m) * KD;

    #pragma unroll
    for (int k0 = 0; k0 < KD; k0 += 32) {
        if (k0 + 32 < KD) {
            __builtin_prefetch(arow0 + k0 + 32, 0, 0);
            __builtin_prefetch(arow1 + k0 + 32, 0, 0);
        }
        v16bf bv;
        #pragma unroll
        for (int j = 0; j < 16; ++j)
            bv[j] = (__bf16)W[(size_t)(k0 + kb + j + (j >= 8 ? 8 : 0)) * Nc + col0 + n];

        v16bf av0 = load_a_bf16(arow0 + k0 + kb);
        acc0 = __builtin_amdgcn_wmma_f32_16x16x32_bf16(
            false, av0, false, bv, (short)0, acc0, false, false);
        v16bf av1 = load_a_bf16(arow1 + k0 + kb);
        acc1 = __builtin_amdgcn_wmma_f32_16x16x32_bf16(
            false, av1, false, bv, (short)0, acc1, false, false);
    }

    const float bb = bias ? bias[col0 + n] : 0.0f;
    const int mrow = half * 8;
    #pragma unroll
    for (int r = 0; r < 8; ++r)
        C[(size_t)(row0 + mrow + r) * Nc + col0 + n] = acc0[r] + bb;
    #pragma unroll
    for (int r = 0; r < 8; ++r)
        C[(size_t)(row0 + 16 + mrow + r) * Nc + col0 + n] = acc1[r] + bb;
}

// ---------------------------------------------------------------------------
// Edge pass 1: logits. One wave per edge; We (32x256, 32KB) + att staged in LDS.
// ef computed on the fly (never materialized: saves 300MB of HBM traffic).
// ---------------------------------------------------------------------------
__global__ __launch_bounds__(256)
void k_edge_logits(const int* __restrict__ srcA, const int* __restrict__ dstA,
                   const float* __restrict__ eattr, const float* __restrict__ loopattr,
                   const float* __restrict__ We, const float* __restrict__ att,
                   const float* __restrict__ xl, const float* __restrict__ xr,
                   float* __restrict__ logits, float* __restrict__ mx,
                   int E, int EA, int heads) {
    __shared__ float sWe[32 * F1C];
    __shared__ float sAtt[F1C];
    for (int i = threadIdx.x; i < 32 * F1C; i += 256) sWe[i] = We[i];
    if (threadIdx.x < F1C) sAtt[threadIdx.x] = att[threadIdx.x];
    __syncthreads();

    const int e = blockIdx.x * 8 + (threadIdx.x >> 5);
    if (e >= EA) return;
    const int lane = threadIdx.x & 31;
    const int s = srcA[e], d = dstA[e];

    const float* arow = (e < E) ? (eattr + (size_t)e * 32)
                                : (loopattr + (size_t)(e - E) * 32);
    const float av = arow[lane];

    const int c0 = lane * 8;   // lanes 0..15 -> channels 0..127 (head 0)
    float ef[8] = {0, 0, 0, 0, 0, 0, 0, 0};
    for (int dd = 0; dd < 32; ++dd) {
        float ad = __shfl(av, dd, 32);
        #pragma unroll
        for (int j = 0; j < 8; ++j) ef[j] += ad * sWe[dd * F1C + c0 + j];
    }

    const float* xls = xl + (size_t)s * F1C + c0;
    const float* xrd = xr + (size_t)d * F1C + c0;
    float4 l0 = *(const float4*)(xls);
    float4 l1 = *(const float4*)(xls + 4);
    float4 r0 = *(const float4*)(xrd);
    float4 r1 = *(const float4*)(xrd + 4);
    float xm[8] = {l0.x + r0.x, l0.y + r0.y, l0.z + r0.z, l0.w + r0.w,
                   l1.x + r1.x, l1.y + r1.y, l1.z + r1.z, l1.w + r1.w};

    float part = 0.0f;
    #pragma unroll
    for (int j = 0; j < 8; ++j) {
        float mv = xm[j] + ef[j];
        float lr = mv > 0.0f ? mv : 0.2f * mv;     // leaky_relu(0.2)
        part += lr * sAtt[c0 + j];
    }
    part += __shfl_xor(part, 1, 32);
    part += __shfl_xor(part, 2, 32);
    part += __shfl_xor(part, 4, 32);
    part += __shfl_xor(part, 8, 32);
    if (heads == 1) {
        part += __shfl_xor(part, 16, 32);
        if (lane == 0) { logits[e] = part; atomicMaxF(&mx[d], part); }
    } else {
        if ((lane & 15) == 0) {
            int h = lane >> 4;
            logits[(size_t)e * 2 + h] = part;
            atomicMaxF(&mx[(size_t)d * 2 + h], part);
        }
    }
}

// Edge pass 2: ex = exp(logit - mx[dst]); denom accumulation.
__global__ void k_exp_den(float* __restrict__ logits, const int* __restrict__ dstA,
                          const float* __restrict__ mx, float* __restrict__ den,
                          int EA, int heads) {
    int t = blockIdx.x * blockDim.x + threadIdx.x;
    if (t >= EA * heads) return;
    int e = t / heads, h = t - e * heads;
    int d = dstA[e];
    float ex = __expf(logits[t] - mx[d * heads + h]);
    logits[t] = ex;
    atomicAdd(&den[d * heads + h], ex);
}

// Edge pass 3: out[dst] += alpha * xl[src]   (thread per edge-channel)
__global__ void k_aggregate(const float* __restrict__ ex, const int* __restrict__ srcA,
                            const int* __restrict__ dstA, const float* __restrict__ den,
                            const float* __restrict__ xl, float* __restrict__ out,
                            int EA, int heads, int ch) {
    size_t t = (size_t)blockIdx.x * blockDim.x + threadIdx.x;
    if (t >= (size_t)EA * F1C) return;
    int e = (int)(t >> 8), c = (int)(t & 255);
    int h = c / ch;
    int s = srcA[e], d = dstA[e];
    float alpha = ex[(size_t)e * heads + h] / den[(size_t)d * heads + h];
    atomicAdd(&out[(size_t)d * F1C + c], alpha * xl[(size_t)s * F1C + c]);
}

__global__ void k_bias_act(float* __restrict__ h, const float* __restrict__ bias,
                           int n, int do_relu) {
    size_t t = (size_t)blockIdx.x * blockDim.x + threadIdx.x;
    if (t >= (size_t)n * F1C) return;
    float v = h[t] + bias[t & 255];
    h[t] = do_relu ? fmaxf(v, 0.0f) : v;
}

// SAGPooling scorer aggregation on ORIGINAL edges: agg[dst] += h[src]
__global__ void k_sag_agg(const int* __restrict__ ei, const float* __restrict__ h,
                          float* __restrict__ agg, int E) {
    size_t t = (size_t)blockIdx.x * blockDim.x + threadIdx.x;
    if (t >= (size_t)E * F1C) return;
    int e = (int)(t >> 8), c = (int)(t & 255);
    int s = ei[e], d = ei[E + e];
    atomicAdd(&agg[(size_t)d * F1C + c], h[(size_t)s * F1C + c]);
}

// score = tanh(agg@Wrel + brel + h@Wroot)  (one wave per node)
__global__ __launch_bounds__(256)
void k_score(const float* __restrict__ agg, const float* __restrict__ h,
             const float* __restrict__ Wrel, const float* __restrict__ brel,
             const float* __restrict__ Wroot, float* __restrict__ score, int n) {
    int node = blockIdx.x * 8 + (threadIdx.x >> 5);
    if (node >= n) return;
    int lane = threadIdx.x & 31;
    float p = 0.0f;
    const float* ar = agg + (size_t)node * F1C;
    const float* hr = h + (size_t)node * F1C;
    for (int c = lane; c < F1C; c += 32) p += ar[c] * Wrel[c] + hr[c] * Wroot[c];
    p += __shfl_xor(p, 1, 32);
    p += __shfl_xor(p, 2, 32);
    p += __shfl_xor(p, 4, 32);
    p += __shfl_xor(p, 8, 32);
    p += __shfl_xor(p, 16, 32);
    if (lane == 0) score[node] = tanhf(p + brel[0]);
}

// Exact top-K per graph (stable, descending = jax.lax.top_k order) fused with
// the final head: out[g*K+rank] = sigmoid(score * dot(h, Wlin) + blin).
__global__ __launch_bounds__(256)
void k_topk_out(const float* __restrict__ score, const float* __restrict__ h,
                const float* __restrict__ Wlin, const float* __restrict__ blin,
                float* __restrict__ out) {
    __shared__ float s[NGP];
    const int g = blockIdx.x;
    const int i = threadIdx.x;
    const int node = g * NGP + i;
    s[i] = score[node];
    __syncthreads();
    const float mys = s[i];
    int rank = 0;
    for (int j = 0; j < NGP; ++j) {
        float sj = s[j];
        rank += (sj > mys) || (sj == mys && j < i);
    }
    if (rank < KEEP) {
        const float* hr = h + (size_t)node * F1C;
        float dot = 0.0f;
        for (int c = 0; c < F1C; ++c) dot += hr[c] * Wlin[c];
        float v = mys * dot + blin[0];
        out[g * KEEP + rank] = 1.0f / (1.0f + __expf(-v));
    }
}

// ---------------------------------------------------------------------------
// launcher
// ---------------------------------------------------------------------------
static inline void fill(float* p, float v, size_t n, hipStream_t st) {
    k_fill<<<(unsigned)((n + 255) / 256), 256, 0, st>>>(p, v, n);
}

extern "C" void kernel_launch(void* const* d_in, const int* in_sizes, int n_in,
                              void* d_out, int out_size, void* d_ws, size_t ws_size,
                              hipStream_t stream) {
    const int N = NNODES, E = NEDGES, EA = NEA;

    const float* x      = (const float*)d_in[0];
    const int*   ei     = (const int*)  d_in[1];   // [2,E]: src then dst
    const float* eattr  = (const float*)d_in[2];
    const float* Wl1    = (const float*)d_in[4];
    const float* bl1    = (const float*)d_in[5];
    const float* Wr1    = (const float*)d_in[6];
    const float* br1    = (const float*)d_in[7];
    const float* We1    = (const float*)d_in[8];
    const float* att1   = (const float*)d_in[9];
    const float* bout1  = (const float*)d_in[10];
    const float* Wl2    = (const float*)d_in[11];
    const float* bl2    = (const float*)d_in[12];
    const float* Wr2    = (const float*)d_in[13];
    const float* br2    = (const float*)d_in[14];
    const float* We2    = (const float*)d_in[15];
    const float* att2   = (const float*)d_in[16];
    const float* bout2  = (const float*)d_in[17];
    const float* Wrel   = (const float*)d_in[18];
    const float* brel   = (const float*)d_in[19];
    const float* Wroot  = (const float*)d_in[20];
    const float* Wlin   = (const float*)d_in[21];
    const float* blin   = (const float*)d_in[22];
    float* out = (float*)d_out;

    // workspace arena (floats)
    float* ws = (float*)d_ws;
    size_t o = 0;
    float* xl  = ws + o; o += (size_t)N * F1C;
    float* xr  = ws + o; o += (size_t)N * F1C;
    float* h1  = ws + o; o += (size_t)N * F1C;
    float* h2  = ws + o; o += (size_t)N * F1C;
    float* la  = ws + o; o += (size_t)N * 32;     // loop attrs (sum then mean)
    float* cnt = ws + o; o += N;
    float* mx  = ws + o; o += (size_t)N * 2;
    float* den = ws + o; o += (size_t)N * 2;
    float* lg  = ws + o; o += (size_t)EA * 2;     // logits -> exp in place
    float* sc  = ws + o; o += N;
    int* srcA  = (int*)(ws + o); o += EA;
    int* dstA  = (int*)(ws + o); o += EA;
    float* agg = xl;                              // reuse after conv2 aggregation

    // --- self loops (fill_value='mean') + augmented edge list ---
    fill(cnt, 0.0f, N, stream);
    fill(la, 0.0f, (size_t)N * 32, stream);
    k_loop_accum<<<(E * 32 + 255) / 256, 256, 0, stream>>>(ei + E, eattr, la, cnt, E);
    k_loop_div<<<(N * 32 + 255) / 256, 256, 0, stream>>>(la, cnt, N);
    k_build_edges<<<(EA + 255) / 256, 256, 0, stream>>>(ei, srcA, dstA, E, EA);

    dim3 blk(256);
    dim3 gg(F1C / 128, N / 32);
    // --- conv1: GATv2(64 -> 2x128) ---
    k_wmma_gemm_bias<64><<<gg, blk, 0, stream>>>(x, Wl1, bl1, xl, N, F1C);
    k_wmma_gemm_bias<64><<<gg, blk, 0, stream>>>(x, Wr1, br1, xr, N, F1C);
    fill(mx, -3.0e38f, (size_t)N * 2, stream);
    fill(den, 0.0f, (size_t)N * 2, stream);
    fill(h1, 0.0f, (size_t)N * F1C, stream);
    k_edge_logits<<<(EA + 7) / 8, blk, 0, stream>>>(srcA, dstA, eattr, la, We1, att1,
                                                    xl, xr, lg, mx, E, EA, 2);
    k_exp_den<<<(EA * 2 + 255) / 256, blk, 0, stream>>>(lg, dstA, mx, den, EA, 2);
    k_aggregate<<<(unsigned)(((size_t)EA * F1C + 255) / 256), blk, 0, stream>>>(
        lg, srcA, dstA, den, xl, h1, EA, 2, 128);
    k_bias_act<<<(unsigned)(((size_t)N * F1C + 255) / 256), blk, 0, stream>>>(h1, bout1, N, 1);

    // --- conv2: GATv2(256 -> 256, heads=1) ---
    k_wmma_gemm_bias<256><<<gg, blk, 0, stream>>>(h1, Wl2, bl2, xl, N, F1C);
    k_wmma_gemm_bias<256><<<gg, blk, 0, stream>>>(h1, Wr2, br2, xr, N, F1C);
    fill(mx, -3.0e38f, N, stream);
    fill(den, 0.0f, N, stream);
    fill(h2, 0.0f, (size_t)N * F1C, stream);
    k_edge_logits<<<(EA + 7) / 8, blk, 0, stream>>>(srcA, dstA, eattr, la, We2, att2,
                                                    xl, xr, lg, mx, E, EA, 1);
    k_exp_den<<<(EA + 255) / 256, blk, 0, stream>>>(lg, dstA, mx, den, EA, 1);
    k_aggregate<<<(unsigned)(((size_t)EA * F1C + 255) / 256), blk, 0, stream>>>(
        lg, srcA, dstA, den, xl, h2, EA, 1, 256);
    k_bias_act<<<(unsigned)(((size_t)N * F1C + 255) / 256), blk, 0, stream>>>(h2, bout2, N, 0);

    // --- SAGPooling scorer (GraphConv on original edges) + top-k + head ---
    fill(agg, 0.0f, (size_t)N * F1C, stream);
    k_sag_agg<<<(unsigned)(((size_t)E * F1C + 255) / 256), blk, 0, stream>>>(ei, h2, agg, E);
    k_score<<<(N + 7) / 8, blk, 0, stream>>>(agg, h2, Wrel, brel, Wroot, sc, N);
    k_topk_out<<<NB, NGP, 0, stream>>>(sc, h2, Wlin, blin, out);
}
